// ContextualConv2d_69389491634228
// MI455X (gfx1250) — matrix operand order repaired and checked by reading, not compiled, not run on hardware
//
#include <hip/hip_runtime.h>

// ---------------------------------------------------------------------------
// ContextualConv2d for MI455X (gfx1250, wave32, WMMA).
//
// Roofline: 77.3 GFLOP conv vs 134 MB HBM traffic -> AI ~577 FLOP/B, compute
// bound by ~5x even at bf16 WMMA rates. v_wmma_f32_16x16x32_bf16 does 16384
// FLOP/instr (8x the f32 16x16x4 WMMA): convert x/w to bf16 once, implicit
// GEMM with f32 accumulation. x(bf16)=32MB + w(bf16)=1.2MB are L2-resident.
//
// v3: - x halo-slab in LDS (4 rows x 66 cols x 32 ci) staged ONCE per
//       ci-chunk; all 9 taps read it via an LDS offset -> 4.4x fewer A
//       global loads, zero A-staging work in the steady-state tap loop.
//     - branch-free zero padding: clamp address into the image, AND the
//       loaded bf16 bits with a precomputed 0/~0 mask (no exec-mask branches,
//       loads can clause).
//     - B tiles ping-pong per flat step s (parity s&1), slab regs for chunk
//       cc+1 prefetched behind 72 WMMAs of chunk cc. One barrier per step.
// ---------------------------------------------------------------------------

typedef __bf16 bf16;
typedef __attribute__((ext_vector_type(16))) __bf16 v16bf;
typedef __attribute__((ext_vector_type(8)))  __bf16 v8bf;
typedef __attribute__((ext_vector_type(8)))  float  v8f;

#define CIN   256
#define COUT  256
#define HW    64
#define IMG   (HW * HW)        // 4096
#define NBATCH 16
#define NPIX  (NBATCH * IMG)   // 65536
#define KTOT  (CIN * 9)        // 2304
#define CDIM  128
#define HDIM  512

#define BPIX 128               // block tile: 128 pixels (2 image rows)
#define BCO  128               //           x 128 output channels
#define BK   32                // K-step (ci chunk)
#define LDA  40                // padded LDS row stride (bf16): 32 data + 8 pad
#define NCC  (CIN / BK)        // 8 ci chunks
#define NSTEP (NCC * 9)        // 72 flat steps
#define SLABS 264              // 4 rows x 66 cols

// ---------------- prep kernels ----------------

__global__ void cvt_x_kernel(const float* __restrict__ x, bf16* __restrict__ xb, int n) {
    int i = blockIdx.x * blockDim.x + threadIdx.x;
    if (i < n) xb[i] = (bf16)x[i];
}

// weight[co][ci][kh][kw] (f32) -> wb[co][tap*256 + ci] (bf16), tap = kh*3+kw
__global__ void cvt_w_kernel(const float* __restrict__ w, bf16* __restrict__ wb) {
    int i = blockIdx.x * blockDim.x + threadIdx.x;
    if (i < COUT * KTOT) {
        int co  = i / KTOT;
        int k   = i - co * KTOT;
        int tap = k >> 8;
        int ci  = k & 255;
        wb[i] = (bf16)w[(co * CIN + ci) * 9 + tap];
    }
}

__global__ void mlp1_kernel(const float* __restrict__ c, const float* __restrict__ w1,
                            const float* __restrict__ b1, float* __restrict__ h) {
    int i = blockIdx.x * blockDim.x + threadIdx.x;
    if (i < NBATCH * HDIM) {
        int n = i >> 9, j = i & (HDIM - 1);
        float s = b1[j];
        for (int k = 0; k < CDIM; ++k) s += c[n * CDIM + k] * w1[j * CDIM + k];
        h[i] = fmaxf(s, 0.0f);
    }
}

__global__ void mlp2_kernel(const float* __restrict__ h, const float* __restrict__ w2,
                            const float* __restrict__ b2, float* __restrict__ ctx) {
    int i = blockIdx.x * blockDim.x + threadIdx.x;
    if (i < NBATCH * COUT) {
        int n = i >> 8, co = i & (COUT - 1);
        float s = b2[co];
        for (int j = 0; j < HDIM; ++j) s += h[n * HDIM + j] * w2[co * HDIM + j];
        ctx[i] = s;
    }
}

// ---------------- main implicit-GEMM conv kernel ----------------

__global__ __launch_bounds__(256)
void conv_wmma_kernel(const bf16* __restrict__ xb, const bf16* __restrict__ wb,
                      const float* __restrict__ ctx, const float* __restrict__ bias,
                      float* __restrict__ out) {
    __shared__ alignas(16) bf16 Xs[SLABS * LDA];     // x halo slab (21.1 KB)
    __shared__ alignas(16) bf16 Bs[2][BCO * LDA];    // ping-pong B tiles (20.5 KB)

    const int t    = threadIdx.x;
    const int lane = t & 31;
    const int wave = t >> 5;

    const int pixBlk = blockIdx.x * BPIX;
    const int coBlk  = blockIdx.y * BCO;

    const int n0 = pixBlk >> 12;            // image index (block constant)
    const int h0 = (pixBlk >> 6) & 63;      // first of the 2 pixel rows (even)

    const unsigned short* ximg = (const unsigned short*)xb + ((size_t)n0 * CIN) * IMG;

    // ---- slab staging geometry (all hoisted, branch-free padding) ----
    // Slot s = r*66 + c covers x[h0-1+r][c-1], r in 0..3, c in 0..65.
    // Thread t owns slot t (x 32 ci), plus exactly one halo element of
    // slot 256+(t>>5) at ci-offset (t&31).
    const int r0  = t / 66, c0 = t - r0 * 66;
    const int hh0 = h0 - 1 + r0, ww0 = c0 - 1;
    const unsigned mask0 =
        (((unsigned)hh0 < 64u) & ((unsigned)ww0 < 64u)) ? 0xFFFFFFFFu : 0u;
    const int hc0 = hh0 < 0 ? 0 : (hh0 > 63 ? 63 : hh0);
    const int wc0 = ww0 < 0 ? 0 : (ww0 > 63 ? 63 : ww0);
    const unsigned short* xs0 = ximg + hc0 * HW + wc0;

    const int slot1 = 256 + (t >> 5);       // r=3, c=58..65
    const int kk1   = t & 31;
    const int ww1   = (slot1 - 3 * 66) - 1; // 57..64
    const int hh1   = h0 + 2;
    const unsigned short mask1 =
        (((unsigned)hh1 < 64u) & ((unsigned)ww1 < 64u)) ? (unsigned short)0xFFFFu
                                                        : (unsigned short)0;
    const int hc1 = hh1 > 63 ? 63 : hh1;
    const int wc1 = ww1 > 63 ? 63 : ww1;
    const unsigned short* xs1 = ximg + (size_t)kk1 * IMG + hc1 * HW + wc1;

    // ---- B staging geometry ----
    const int coT = t >> 3;                 // 0..31
    const int k4  = (t & 7) << 2;           // 0..28
    const bf16* wsrc = wb + (size_t)(coBlk + coT) * KTOT + k4;

    const int m  = lane & 15;
    const int hk = lane >> 4;

    // Per-p A fragment base offset in the slab (pixel geometry, hoisted).
    int abase[2];
    #pragma unroll
    for (int p = 0; p < 2; ++p) {
        int rowp = (wave & 3) * 32 + p * 16 + m;      // pixel-local 0..127
        int hpp  = rowp >> 6, wp = rowp & 63;
        abase[p] = ((hpp + 1) * 66 + (wp + 1)) * LDA + hk * 8;
    }

    v8f acc[2][4];
    #pragma unroll
    for (int p = 0; p < 2; ++p)
        #pragma unroll
        for (int cg = 0; cg < 4; ++cg) acc[p][cg] = v8f{};

    alignas(16) unsigned short sa[32];      // staged slab slice (slot t)
    unsigned short sa1;                     // staged halo element
    uint2 rb[4];                            // staged B

    auto stageSlab = [&](int cc) {          // global -> regs, 33 loads
        const size_t coff = (size_t)(cc * BK) * IMG;
        #pragma unroll
        for (int i = 0; i < 32; ++i)
            sa[i] = xs0[coff + (size_t)i * IMG];
        sa1 = xs1[coff];
    };
    auto commitSlab = [&]() {               // regs -> LDS, masked, 4x b128 + b16
        #pragma unroll
        for (int j = 0; j < 4; ++j) {
            uint4 q = *(const uint4*)&sa[j * 8];
            q.x &= mask0; q.y &= mask0; q.z &= mask0; q.w &= mask0;
            *(uint4*)&Xs[t * LDA + j * 8] = q;
        }
        ((unsigned short*)Xs)[slot1 * LDA + kk1] = (unsigned short)(sa1 & mask1);
    };
    auto stageB = [&](int chunk, int tapIdx) {
        const bf16* pb = wsrc + tapIdx * 256 + chunk * BK;
        #pragma unroll
        for (int i = 0; i < 4; ++i)
            rb[i] = *(const uint2*)(pb + (size_t)(32 * i) * KTOT);
    };
    auto commitB = [&](int d) {
        #pragma unroll
        for (int i = 0; i < 4; ++i)
            *(uint2*)&Bs[d][(coT + 32 * i) * LDA + k4] = rb[i];
    };

    auto computeTap = [&](int dh, int dw, int d) {
        const int doff = (dh * 66 + dw) * LDA;
        // A frag (16-bit A 16x32): lane l -> M=l&15; elems 0..7 -> K=(l>>4)*8+i,
        // elems 8..15 -> K=16+(l>>4)*8+i.
        v16bf afr[2];
        #pragma unroll
        for (int p = 0; p < 2; ++p) {
            const bf16* pa = &Xs[abase[p] + doff];
            v8bf lo = *(const v8bf*)pa;
            v8bf hi = *(const v8bf*)(pa + 16);
            afr[p] = __builtin_shufflevector(lo, hi,
                0,1,2,3,4,5,6,7,8,9,10,11,12,13,14,15);
        }
        // B frag (16-bit B 32x16): lane l -> N=l&15; elems 0..7 -> K=(l>>4)*8+i, elems 8..15 -> K=16+(l>>4)*8+i.
        v16bf bfr[4];
        #pragma unroll
        for (int cg = 0; cg < 4; ++cg) {
            const bf16* pb = &Bs[d][((wave >> 2) * 64 + cg * 16 + m) * LDA + hk * 8];
            v8bf lo = *(const v8bf*)pb;
            v8bf hi = *(const v8bf*)(pb + 16);
            bfr[cg] = __builtin_shufflevector(lo, hi,
                0,1,2,3,4,5,6,7,8,9,10,11,12,13,14,15);
        }
        #pragma unroll
        for (int p = 0; p < 2; ++p)
            #pragma unroll
            for (int cg = 0; cg < 4; ++cg)
                acc[p][cg] = __builtin_amdgcn_wmma_f32_16x16x32_bf16(
                    false, afr[p], false, bfr[cg],
                    (short)0, acc[p][cg], false, false);
    };

    // ---- pipelined main loop ----
    stageSlab(0);
    stageB(0, 0);
    commitSlab();
    commitB(0);
    stageSlab(1);                       // prefetch chunk 1 behind chunk 0 taps
    __syncthreads();

    for (int cc = 0; cc < NCC; ++cc) {
        if (cc > 0) {
            commitSlab();               // slab for cc (prefetched during cc-1)
            if (cc + 1 < NCC) stageSlab(cc + 1);
            __syncthreads();
        }
        #pragma unroll
        for (int tap = 0; tap < 9; ++tap) {
            const int s = cc * 9 + tap;
            if (s + 1 < NSTEP) {
                if (tap == 8) stageB(cc + 1, 0);
                else          stageB(cc, tap + 1);
            }
            computeTap(tap / 3 - 1, tap % 3 - 1, s & 1);
            if (s + 1 < NSTEP) commitB((s + 1) & 1);
            __syncthreads();
        }
    }

    // Epilogue: out[n][co][h][w] = acc + ctx[n][co] + bias[co]
    // C/D layout: lane l -> N = l&15 ; VGPR v -> M = v + 8*(l>>4)
    const int wPix = pixBlk + (wave & 3) * 32;
    const int wCo  = coBlk  + (wave >> 2) * 64;
    #pragma unroll
    for (int p = 0; p < 2; ++p) {
        int pix0 = wPix + p * 16 + (hk << 3);
        #pragma unroll
        for (int cg = 0; cg < 4; ++cg) {
            int co = wCo + cg * 16 + m;
            float addv = ctx[n0 * COUT + co] + bias[co];
            #pragma unroll
            for (int v = 0; v < 8; ++v) {
                int opix = pix0 + v;
                out[(size_t)n0 * (COUT * IMG) + (size_t)co * IMG + (opix & (IMG - 1))]
                    = acc[p][cg][v] + addv;
            }
        }
    }
}

// ---------------- launcher ----------------

extern "C" void kernel_launch(void* const* d_in, const int* in_sizes, int n_in,
                              void* d_out, int out_size, void* d_ws, size_t ws_size,
                              hipStream_t stream) {
    const float* x    = (const float*)d_in[0];  // (16,256,64,64)
    const float* c    = (const float*)d_in[1];  // (16,128)
    const float* wgt  = (const float*)d_in[2];  // (256,256,3,3)
    const float* bias = (const float*)d_in[3];  // (256,)
    const float* w1   = (const float*)d_in[4];  // (512,128)
    const float* b1   = (const float*)d_in[5];  // (512,)
    const float* w2   = (const float*)d_in[6];  // (256,512)
    const float* b2   = (const float*)d_in[7];  // (256,)
    float* out = (float*)d_out;

    // Workspace layout (all 256B aligned):
    //   xb : 33,554,432 B ; wb : 1,179,648 B ; h : 32,768 B ; ctx : 16,384 B
    char* ws = (char*)d_ws;
    bf16*  xb  = (bf16*)(ws);
    bf16*  wb  = (bf16*)(ws + 33554432);
    float* h   = (float*)(ws + 33554432 + 1179648);
    float* ctx = (float*)(ws + 33554432 + 1179648 + 32768);

    const int nX = NBATCH * CIN * IMG;   // 16,777,216
    cvt_x_kernel<<<(nX + 255) / 256, 256, 0, stream>>>(x, xb, nX);
    cvt_w_kernel<<<(COUT * KTOT + 255) / 256, 256, 0, stream>>>(wgt, wb);
    mlp1_kernel<<<(NBATCH * HDIM + 255) / 256, 256, 0, stream>>>(c, w1, b1, h);
    mlp2_kernel<<<(NBATCH * COUT + 255) / 256, 256, 0, stream>>>(h, w2, b2, ctx);

    dim3 grid(NPIX / BPIX, COUT / BCO);  // (512, 2)
    conv_wmma_kernel<<<grid, 256, 0, stream>>>(xb, wb, ctx, bias, out);
}
